// CausalAttention_55877524521082
// MI455X (gfx1250) — compile-verified, matrix operand order checked by reading, not run on hardware
//
#include <hip/hip_runtime.h>
#include <math.h>

typedef __attribute__((ext_vector_type(16))) _Float16 v16h;
typedef __attribute__((ext_vector_type(8)))  _Float16 v8h;
typedef __attribute__((ext_vector_type(8)))  float    v8f;
typedef __attribute__((ext_vector_type(4)))  float    f32x4;
typedef __attribute__((ext_vector_type(4)))  unsigned int u32x4;
typedef __attribute__((ext_vector_type(4)))  int      i32x4;
typedef __attribute__((ext_vector_type(8)))  int      i32x8;

#define B_   4
#define S_   4096
#define DIN  512
#define DOUT 64
#define ROWS (B_ * S_)          // 16384

// Packed weights: kc-major tiles. tile = kc*12 + m*4 + nt, 512 halves each.
// Stage g = kc in [4g, 4g+3] -> 48 tiles = 24576 halves = 48 KB contiguous.
#define WP_TILES      192
#define WP_HALVES     (WP_TILES * 512)     // 98304
#define STAGE_HALVES  24576
#define STAGE_DWORDS  12288

// ---------------------------------------------------------------------------
// WMMA lane-layout helpers (wave32, 16x16x32 f16 -> f32).
// A (16x32): lane L: row M=L%16, h=L/16, element e: K=(e<8?0:16)+8h+(e&7)
// B (32x16): lane L: col N=L%16, same K pattern.
// C/D (16x16 f32): lane L: col N=L%16; VGPR r: row M=r+8h.
// ---------------------------------------------------------------------------
static __device__ inline v16h load_tile_f16(const _Float16* p) {
    v16h r;
    *reinterpret_cast<v8h*>(&r)     = *reinterpret_cast<const v8h*>(p);
    *(reinterpret_cast<v8h*>(&r)+1) = *reinterpret_cast<const v8h*>(p + 16);
    return r;
}

static __device__ inline v8f wmma_f16(v16h a, v16h b, v8f c) {
    return __builtin_amdgcn_wmma_f32_16x16x32_f16(false, a, false, b,
                                                  (short)0, c, false, false);
}

// ---------------------------------------------------------------------------
// Kernel 0: repack Wq/Wk/Wv (f32 [512][64]) -> f16 WMMA-B lane layout, kc-major.
// ---------------------------------------------------------------------------
__global__ void __launch_bounds__(32)
pack_w_kernel(const float* __restrict__ Wq, const float* __restrict__ Wk,
              const float* __restrict__ Wv, _Float16* __restrict__ wp) {
    const int tile = blockIdx.x;            // 0..191 = kc*12 + m*4 + nt
    const int kc   = tile / 12;
    const int m    = (tile / 4) % 3;
    const int nt   = tile & 3;
    const float* W = (m == 0) ? Wq : ((m == 1) ? Wk : Wv);

    const int lane = threadIdx.x & 31;
    const int h    = lane >> 4;
    const int n    = nt * 16 + (lane & 15);

    _Float16* dst = wp + (size_t)tile * 512 + lane * 16;
#pragma unroll
    for (int e = 0; e < 16; ++e) {
        int k = kc * 32 + ((e < 8) ? 0 : 16) + h * 8 + (e & 7);
        dst[e] = (_Float16)W[k * DOUT + n];
    }
}

// ---------------------------------------------------------------------------
// TDM: async-load one contiguous 48 KB weight stage into LDS.
// D# group0: count=1 | lds_addr | global_addr | type=2
// D# group1: data_size=4B; tensor_dim0=tile_dim0=12288 dwords; dims1=1.
// ---------------------------------------------------------------------------
static __device__ inline void tdm_load_stage(const _Float16* src, unsigned ldsOff) {
#if __has_builtin(__builtin_amdgcn_tensor_load_to_lds)
    const unsigned long long ga = (unsigned long long)(const void*)src;
    u32x4 g0 = { 1u,                                   // count=1, user desc
                 ldsOff,                               // LDS byte address
                 (unsigned)ga,                         // global addr [95:64]
                 (unsigned)(ga >> 32) | (2u << 30) };  // addr msbs | type=2
    i32x8 g1 = { (int)(2u << 16),                      // data_size = 4 bytes
                 (int)((STAGE_DWORDS & 0xFFFF) << 16), // tensor_dim0 lo
                 (int)((STAGE_DWORDS >> 16) | (1 << 16)), // dim0 hi | tensor_dim1=1
                 (int)((STAGE_DWORDS & 0xFFFF) << 16), // tile_dim0
                 1,                                    // tile_dim1=1
                 STAGE_DWORDS,                         // tensor_dim0_stride
                 0, 0 };
    i32x4 z4 = { 0, 0, 0, 0 };
#if defined(__clang_major__) && (__clang_major__ >= 23)
    i32x8 z8 = { 0, 0, 0, 0, 0, 0, 0, 0 };
    __builtin_amdgcn_tensor_load_to_lds(g0, g1, z4, z4, z8, 0);
#else
    __builtin_amdgcn_tensor_load_to_lds(g0, g1, z4, z4, 0);
#endif
    __builtin_amdgcn_s_wait_tensorcnt(0);
#endif
}

// ---------------------------------------------------------------------------
// Kernel 1: fused Q/K/V projection GEMM, weights staged in LDS via TDM.
// wave -> 16 rows of x; acc[3][4] = 16x64 f32 per matrix.
// Q,K row-major f16 [ROWS][64]; V transposed [B][64][S] f16; Q pre-scaled.
// ---------------------------------------------------------------------------
__global__ void __launch_bounds__(128)
proj_kernel(const float* __restrict__ x, const _Float16* __restrict__ wp,
            _Float16* __restrict__ qh, _Float16* __restrict__ kh,
            _Float16* __restrict__ vth) {
    __shared__ _Float16 lws[STAGE_HALVES];   // 48 KB

    const int wave = (blockIdx.x * blockDim.x + threadIdx.x) >> 5;  // 0..1023
    const int lane = threadIdx.x & 31;
    const int h    = lane >> 4;
    const int ln   = lane & 15;
    const int row0 = wave * 16;

    v8f acc[3][4];
#pragma unroll
    for (int m = 0; m < 3; ++m)
#pragma unroll
        for (int nt = 0; nt < 4; ++nt) acc[m][nt] = {};

    for (int g = 0; g < 4; ++g) {
        const _Float16* src = wp + (size_t)g * STAGE_HALVES;
#if __has_builtin(__builtin_amdgcn_tensor_load_to_lds)
        if ((threadIdx.x >> 5) == 0)   // wave 0 drives the TDM, then waits
            tdm_load_stage(src, (unsigned)(size_t)&lws[0]);
#else
        for (int i = threadIdx.x; i < STAGE_HALVES / 8; i += 128)
            reinterpret_cast<v8h*>(lws)[i] =
                reinterpret_cast<const v8h*>(src)[i];
#endif
        __syncthreads();

#pragma unroll
        for (int kk = 0; kk < 4; ++kk) {
            const int kc = g * 4 + kk;
            // A tile: 16 rows x 32 dims of x, f32 -> f16
            const float* xr = x + (size_t)(row0 + ln) * DIN + kc * 32 + h * 8;
            const f32x4* xp = reinterpret_cast<const f32x4*>(xr);
            f32x4 x0 = xp[0], x1 = xp[1];      // dims +0..7
            f32x4 x2 = xp[4], x3 = xp[5];      // dims +16..23
            v16h a;
#pragma unroll
            for (int j = 0; j < 4; ++j) {
                a[j]      = (_Float16)x0[j];
                a[4 + j]  = (_Float16)x1[j];
                a[8 + j]  = (_Float16)x2[j];
                a[12 + j] = (_Float16)x3[j];
            }
#pragma unroll
            for (int m = 0; m < 3; ++m) {
#pragma unroll
                for (int nt = 0; nt < 4; ++nt) {
                    const _Float16* bp = lws + (size_t)(kk * 12 + m * 4 + nt) * 512
                                             + lane * 16;
                    v16h b = load_tile_f16(bp - 0);   // two ds_load_b128
                    // B layout halves are adjacent here: [0..7],[8..15] packed
                    *reinterpret_cast<v8h*>(&b)     = *reinterpret_cast<const v8h*>(bp);
                    *(reinterpret_cast<v8h*>(&b)+1) = *reinterpret_cast<const v8h*>(bp + 8);
                    acc[m][nt] = wmma_f16(a, b, acc[m][nt]);
                }
            }
        }
        __syncthreads();
    }

    // pre-scale Q by 1/sqrt(DOUT) = 0.125
#pragma unroll
    for (int nt = 0; nt < 4; ++nt)
#pragma unroll
        for (int r = 0; r < 8; ++r) acc[0][nt][r] *= 0.125f;

    const int b  = row0 / S_;
    const int sl = row0 % S_;
#pragma unroll
    for (int nt = 0; nt < 4; ++nt) {
        const int n = nt * 16 + ln;
#pragma unroll
        for (int r = 0; r < 8; ++r) {
            const int rr = r + 8 * h;
            qh[(size_t)(row0 + rr) * DOUT + n] = (_Float16)acc[0][nt][r];
            kh[(size_t)(row0 + rr) * DOUT + n] = (_Float16)acc[1][nt][r];
        }
        v8h vv;
#pragma unroll
        for (int r = 0; r < 8; ++r) vv[r] = (_Float16)acc[2][nt][r];
        *reinterpret_cast<v8h*>(vth + ((size_t)(b * DOUT + n)) * S_ + sl + 8 * h) = vv;
    }
}

// ---------------------------------------------------------------------------
// Kernel 2: fused causal flash attention (transposed-score formulation).
//   S^T(32k x 16q) = K_tile(A) * Q^T(B); D-layout of P^T == B-layout for PV.
//   O^T(64d x 16q) += V^T_tile(A) * P^T(B)
// K tiles double-buffered across iterations; V tiles issued early so softmax
// VALU/trans work hides their latency.
// ---------------------------------------------------------------------------
__global__ void __launch_bounds__(128)
attn_kernel(const _Float16* __restrict__ qh, const _Float16* __restrict__ kh,
            const _Float16* __restrict__ vth, float* __restrict__ out) {
    const int wave = (blockIdx.x * blockDim.x + threadIdx.x) >> 5;  // 0..1023
    const int lane = threadIdx.x & 31;
    const int h    = lane >> 4;
    const int ln   = lane & 15;

    const int b     = wave / (S_ / 16);
    const int q0    = (wave % (S_ / 16)) * 16;
    const int qglob = q0 + ln;

    const _Float16* qrow   = qh + (size_t)(b * S_ + q0 + ln) * DOUT;
    const _Float16* kbatch = kh + (size_t)b * S_ * DOUT;
    const _Float16* vtb    = vth + (size_t)b * DOUT * S_;

    // Q tile as B matrices for the two 32-wide dim chunks
    v16h bq[2];
#pragma unroll
    for (int c = 0; c < 2; ++c) bq[c] = load_tile_f16(qrow + c * 32 + h * 8);

    v8f o[4];
#pragma unroll
    for (int dc = 0; dc < 4; ++dc) o[dc] = {};
    float mrun = -__builtin_inff();
    float lrun = 0.0f;

    const int ntiles = (q0 + 16 + 31) >> 5;

    // K-tile register double buffer: [kt0_c0, kt1_c0, kt0_c1, kt1_c1]
    v16h ak[2][4];
    {
        const int kb = 0;
#pragma unroll
        for (int c = 0; c < 2; ++c) {
            ak[0][2*c+0] = load_tile_f16(kbatch + (size_t)(kb + ln)      * DOUT + c*32 + h*8);
            ak[0][2*c+1] = load_tile_f16(kbatch + (size_t)(kb + 16 + ln) * DOUT + c*32 + h*8);
        }
    }

    for (int t = 0; t < ntiles; ++t) {
        const int kbase = t * 32;
        const int cur   = t & 1;

        // issue V-tile loads early (consumed after softmax)
        v16h av[4];
#pragma unroll
        for (int dc = 0; dc < 4; ++dc)
            av[dc] = load_tile_f16(vtb + (size_t)(dc * 16 + ln) * S_ + kbase + 8 * h);

        // issue next iteration's K tiles
        if (t + 1 < ntiles) {
            const int kb = kbase + 32;
            __builtin_prefetch(kbatch + (size_t)(kb + 32 + ln) * DOUT, 0, 0);
#pragma unroll
            for (int c = 0; c < 2; ++c) {
                ak[cur^1][2*c+0] = load_tile_f16(kbatch + (size_t)(kb + ln)      * DOUT + c*32 + h*8);
                ak[cur^1][2*c+1] = load_tile_f16(kbatch + (size_t)(kb + 16 + ln) * DOUT + c*32 + h*8);
            }
        }

        // S^T tiles
        v8f s0 = {}, s1 = {};
        s0 = wmma_f16(ak[cur][0], bq[0], s0);
        s0 = wmma_f16(ak[cur][2], bq[1], s0);
        s1 = wmma_f16(ak[cur][1], bq[0], s1);
        s1 = wmma_f16(ak[cur][3], bq[1], s1);

        // causal mask: lane holds keys kbase + {8h..8h+7} and +16
#pragma unroll
        for (int r = 0; r < 8; ++r) {
            const int k0i = kbase + 8 * h + r;
            if (k0i > qglob)      s0[r] = -__builtin_inff();
            if (k0i + 16 > qglob) s1[r] = -__builtin_inff();
        }

        // tile max across lane pair
        float tm = s0[0];
#pragma unroll
        for (int r = 1; r < 8; ++r) tm = fmaxf(tm, s0[r]);
#pragma unroll
        for (int r = 0; r < 8; ++r) tm = fmaxf(tm, s1[r]);
        tm = fmaxf(tm, __shfl_xor(tm, 16, 32));

        const float mnew  = fmaxf(mrun, tm);
        const float alpha = __expf(mrun - mnew);

        // P^T: D-layout maps directly onto the next WMMA's B-layout
        float ps = 0.0f;
        v16h bp;
#pragma unroll
        for (int r = 0; r < 8; ++r) {
            float p0 = __expf(s0[r] - mnew);
            float p1 = __expf(s1[r] - mnew);
            ps += p0 + p1;
            bp[r]     = (_Float16)p0;
            bp[8 + r] = (_Float16)p1;
        }
        ps += __shfl_xor(ps, 16, 32);
        lrun = lrun * alpha + ps;
        mrun = mnew;

        // O^T rescale + accumulate
#pragma unroll
        for (int dc = 0; dc < 4; ++dc) {
#pragma unroll
            for (int r = 0; r < 8; ++r) o[dc][r] *= alpha;
            o[dc] = wmma_f16(av[dc], bp, o[dc]);
        }
    }

    // normalize and store: lane writes 8 contiguous f32 per dim-chunk
    const float inv = 1.0f / lrun;
    float* orow = out + (size_t)(b * S_ + q0 + ln) * DOUT;
#pragma unroll
    for (int dc = 0; dc < 4; ++dc) {
        f32x4 u0, u1;
#pragma unroll
        for (int j = 0; j < 4; ++j) {
            u0[j] = o[dc][j] * inv;
            u1[j] = o[dc][4 + j] * inv;
        }
        *reinterpret_cast<f32x4*>(orow + dc * 16 + 8 * h)     = u0;
        *reinterpret_cast<f32x4*>(orow + dc * 16 + 8 * h + 4) = u1;
    }
}

// ---------------------------------------------------------------------------
extern "C" void kernel_launch(void* const* d_in, const int* in_sizes, int n_in,
                              void* d_out, int out_size, void* d_ws, size_t ws_size,
                              hipStream_t stream) {
    const float* x  = (const float*)d_in[0];
    const float* Wq = (const float*)d_in[1];
    const float* Wk = (const float*)d_in[2];
    const float* Wv = (const float*)d_in[3];
    float* out = (float*)d_out;

    // ws layout (f16 units): Wpack | Q | K | V^T
    _Float16* wp  = (_Float16*)d_ws;
    _Float16* qh  = wp + WP_HALVES;
    _Float16* kh  = qh + (size_t)ROWS * DOUT;
    _Float16* vth = kh + (size_t)ROWS * DOUT;

    pack_w_kernel<<<192, 32, 0, stream>>>(Wq, Wk, Wv, wp);
    proj_kernel<<<256, 128, 0, stream>>>(x, wp, qh, kh, vth);
    attn_kernel<<<256, 128, 0, stream>>>(qh, kh, vth, out);
}
